// CrossModalAttention_29068338659454
// MI455X (gfx1250) — matrix operand order repaired and checked by reading, not compiled, not run on hardware
//
#include <hip/hip_runtime.h>
#include <math.h>

// CDNA5 (gfx1250) vector types for WMMA
typedef __attribute__((ext_vector_type(16))) __bf16 v16bf;
typedef __attribute__((ext_vector_type(8)))  float  v8f;

#define HID 128

// ---------------------------------------------------------------------------
// round-to-nearest-even f32 -> bf16 (bit trick, used on host-visible buffers)
__device__ __forceinline__ unsigned short f32_to_bf16_bits(float f) {
    union { float f; unsigned int u; } x; x.f = f;
    unsigned int r = x.u + 0x7FFFu + ((x.u >> 16) & 1u);
    return (unsigned short)(r >> 16);
}

// ---------------------------------------------------------------------------
// Pack W1 [128,256] fp32 (torch [out,in]) into bf16 tiles laid out exactly as
// the CDNA5 WMMA B-operand (32x16 per tile, per-lane 16 contiguous bf16):
//   idx = ((k*8 + n)*32 + lane)*16 + j
//   value = W1[h = n*16 + lane%16][d = k*32 + (lane>=16 ? 8 : 0) + (j&7) + (j>=8 ? 16 : 0)]
__global__ void w1_pack_kernel(const float* __restrict__ W1,
                               unsigned short* __restrict__ w1bf) {
    int t = blockIdx.x * blockDim.x + threadIdx.x;   // 0 .. 32767
    if (t >= 8 * 8 * 32 * 16) return;
    int j = t & 15;
    int L = (t >> 4) & 31;
    int n = (t >> 9) & 7;
    int k = t >> 12;
    int h = n * 16 + (L & 15);
    int d = k * 32 + ((L >> 4) ? 8 : 0) + (j & 7) + ((j >> 3) ? 16 : 0);
    w1bf[t] = f32_to_bf16_bits(W1[h * 256 + d]);
}

// ---------------------------------------------------------------------------
// One wave computes 16 edges: hidden = tanh(attn_in @ W1^T + b1),
// logit = hidden @ W2 + b2, via v_wmma_f32_16x16x32_bf16.
__global__ __launch_bounds__(256) void attn_logits_kernel(
    const float* __restrict__ progF, const float* __restrict__ voxF,
    const int* __restrict__ src, const int* __restrict__ dst,
    const uint4* __restrict__ w1bf,          // 64KB packed bf16 B tiles
    const float* __restrict__ b1, const float* __restrict__ W2,
    const float* __restrict__ b2,
    float* __restrict__ logits, int nTiles, long E) {
    __shared__ __align__(32) uint4 ldsW[4096];           // 64 KB
    for (int i = threadIdx.x; i < 4096; i += blockDim.x) ldsW[i] = w1bf[i];
    __syncthreads();

    const int wave = threadIdx.x >> 5;
    const int lane = threadIdx.x & 31;
    const int tile = blockIdx.x * (blockDim.x >> 5) + wave;
    if (tile >= nTiles) return;                          // wave-uniform: EXEC stays all-1

    const int m    = lane & 15;                          // edge row in A layout, hidden col N in C layout
    const int half = lane >> 4;                          // K-half selector (A layout)
    long e = (long)tile * 16 + m;
    if (e >= E) e = E - 1;                               // clamp (no EXEC holes for WMMA)
    const int si = src[e];
    const int di = dst[e];
    const float* pp = progF + (long)si * HID;
    const float* vp = voxF  + (long)di * HID;

    float b1v[8], w2v[8];
#pragma unroll
    for (int n = 0; n < 8; ++n) { b1v[n] = b1[n * 16 + m]; w2v[n] = W2[n * 16 + m]; }

    v8f acc[8] = {};                                     // 8 N-tiles of 16x16 f32 accum
    const __bf16* ldsB = (const __bf16*)ldsW;

#pragma unroll
    for (int k = 0; k < 8; ++k) {                        // K = 256 in steps of 32
        const float* fp = (k < 4) ? pp : vp;             // concat(prog_e, vox_e)
        const int d0 = (k & 3) * 32 + half * 8;
        float4 f0 = *(const float4*)(fp + d0);
        float4 f1 = *(const float4*)(fp + d0 + 4);
        float4 f2 = *(const float4*)(fp + d0 + 16);
        float4 f3 = *(const float4*)(fp + d0 + 20);
        v16bf a;
        a[0]  = (__bf16)f0.x; a[1]  = (__bf16)f0.y; a[2]  = (__bf16)f0.z; a[3]  = (__bf16)f0.w;
        a[4]  = (__bf16)f1.x; a[5]  = (__bf16)f1.y; a[6]  = (__bf16)f1.z; a[7]  = (__bf16)f1.w;
        a[8]  = (__bf16)f2.x; a[9]  = (__bf16)f2.y; a[10] = (__bf16)f2.z; a[11] = (__bf16)f2.w;
        a[12] = (__bf16)f3.x; a[13] = (__bf16)f3.y; a[14] = (__bf16)f3.z; a[15] = (__bf16)f3.w;
#pragma unroll
        for (int n = 0; n < 8; ++n) {
            v16bf bmat = *(const v16bf*)(ldsB + (((k * 8 + n) * 32 + lane) << 4));
            acc[n] = __builtin_amdgcn_wmma_f32_16x16x32_bf16(
                false, a, false, bmat, (short)0, acc[n], false, false);
        }
    }

    // logit[e] = sum_h tanh(hidden_pre[e,h]) * W2[h] + b2
    // C layout: VGPR r, lane L -> row M = r + 8*half, col N = L%16
    const float b2s = b2[0];
#pragma unroll
    for (int r = 0; r < 8; ++r) {
        float part = 0.f;
#pragma unroll
        for (int n = 0; n < 8; ++n)
            part += tanhf(acc[n][r] + b1v[n]) * w2v[n];
#pragma unroll
        for (int off = 1; off < 16; off <<= 1)           // reduce within each 16-lane half
            part += __shfl_xor(part, off, 16);
        long idx = (long)tile * 16 + half * 8 + r;
        if (m == 0 && idx < E) logits[idx] = part + b2s;
    }
}

// ---------------------------------------------------------------------------
// Global softmax reductions
__global__ void reduce_max_partial(const float* __restrict__ logits, long E,
                                   float* __restrict__ partials) {
    __shared__ float sm[256];
    float v = -INFINITY;
    for (long i = (long)blockIdx.x * blockDim.x + threadIdx.x; i < E;
         i += (long)gridDim.x * blockDim.x)
        v = fmaxf(v, logits[i]);
    sm[threadIdx.x] = v;
    __syncthreads();
    for (int s = 128; s > 0; s >>= 1) {
        if (threadIdx.x < s) sm[threadIdx.x] = fmaxf(sm[threadIdx.x], sm[threadIdx.x + s]);
        __syncthreads();
    }
    if (threadIdx.x == 0) partials[blockIdx.x] = sm[0];
}

__global__ void reduce_max_final(const float* __restrict__ partials, float* __restrict__ gmax) {
    __shared__ float sm[256];
    sm[threadIdx.x] = partials[threadIdx.x];
    __syncthreads();
    for (int s = 128; s > 0; s >>= 1) {
        if (threadIdx.x < s) sm[threadIdx.x] = fmaxf(sm[threadIdx.x], sm[threadIdx.x + s]);
        __syncthreads();
    }
    if (threadIdx.x == 0) gmax[0] = sm[0];
}

__global__ void reduce_sum_partial(const float* __restrict__ logits, long E,
                                   const float* __restrict__ gmax,
                                   float* __restrict__ partials) {
    __shared__ float sm[256];
    const float gm = gmax[0];
    float v = 0.f;
    for (long i = (long)blockIdx.x * blockDim.x + threadIdx.x; i < E;
         i += (long)gridDim.x * blockDim.x)
        v += __expf(logits[i] - gm);
    sm[threadIdx.x] = v;
    __syncthreads();
    for (int s = 128; s > 0; s >>= 1) {
        if (threadIdx.x < s) sm[threadIdx.x] += sm[threadIdx.x + s];
        __syncthreads();
    }
    if (threadIdx.x == 0) partials[blockIdx.x] = sm[0];
}

__global__ void reduce_sum_final(const float* __restrict__ partials, float* __restrict__ gsum) {
    __shared__ float sm[256];
    sm[threadIdx.x] = partials[threadIdx.x];
    __syncthreads();
    for (int s = 128; s > 0; s >>= 1) {
        if (threadIdx.x < s) sm[threadIdx.x] += sm[threadIdx.x + s];
        __syncthreads();
    }
    if (threadIdx.x == 0) gsum[0] = sm[0];
}

// attention_weights[e] = exp(logit - max) / sum  -> tail of d_out
__global__ void weights_kernel(const float* __restrict__ logits, long E,
                               const float* __restrict__ gstats,   // [max, sum]
                               float* __restrict__ outw) {
    long i = (long)blockIdx.x * blockDim.x + threadIdx.x;
    if (i >= E) return;
    outw[i] = __expf(logits[i] - gstats[0]) / gstats[1];
}

// zero output_features region (atomics accumulate into it)
__global__ void zero_kernel(float4* __restrict__ out, long n4) {
    for (long i = (long)blockIdx.x * blockDim.x + threadIdx.x; i < n4;
         i += (long)gridDim.x * blockDim.x)
        out[i] = make_float4(0.f, 0.f, 0.f, 0.f);
}

// scatter-add: out[dst[e]] += w[e] * prog[src[e]]   (32 threads / edge, float4 each)
__global__ void scatter_kernel(const float* __restrict__ progF,
                               const int* __restrict__ src, const int* __restrict__ dst,
                               const float* __restrict__ w, float* __restrict__ out,
                               long E) {
    long t = (long)blockIdx.x * blockDim.x + threadIdx.x;
    long e = t >> 5;
    if (e >= E) return;
    int q = (int)(t & 31);
    float wt = w[e];
    const float4 p = *(const float4*)(progF + (long)src[e] * HID + q * 4);
    float* o = out + (long)dst[e] * HID + q * 4;
    atomicAdd(o + 0, wt * p.x);
    atomicAdd(o + 1, wt * p.y);
    atomicAdd(o + 2, wt * p.z);
    atomicAdd(o + 3, wt * p.w);
}

// ---------------------------------------------------------------------------
extern "C" void kernel_launch(void* const* d_in, const int* in_sizes, int n_in,
                              void* d_out, int out_size, void* d_ws, size_t ws_size,
                              hipStream_t stream) {
    const float* progF = (const float*)d_in[0];
    const float* voxF  = (const float*)d_in[1];
    const int*   eidx  = (const int*)d_in[2];     // [2, E]: row0=src, row1=dst
    const float* W1    = (const float*)d_in[3];
    const float* b1    = (const float*)d_in[4];
    const float* W2    = (const float*)d_in[5];
    const float* b2    = (const float*)d_in[6];

    const long E    = in_sizes[2] / 2;            // 500000
    const long NVOX = in_sizes[1] / HID;          // 200000
    const int* src = eidx;
    const int* dst = eidx + E;

    // workspace layout (floats): [0..1] stats, [2..257] partials,
    // [512 .. 512+E) logits, then 64KB packed bf16 W1 (32B aligned: E%8==0)
    float* ws       = (float*)d_ws;
    float* gstats   = ws;
    float* partials = ws + 2;
    float* logits   = ws + 512;
    unsigned short* w1bf = (unsigned short*)(ws + 512 + E);

    float* out  = (float*)d_out;                  // [NVOX, HID]
    float* outw = out + NVOX * (long)HID;         // [E] attention weights

    // 1. pack W1 -> bf16 B-operand layout
    w1_pack_kernel<<<dim3(128), dim3(256), 0, stream>>>(W1, w1bf);

    // 2. WMMA logits (one wave per 16-edge tile, 8 waves / block)
    long nTiles  = (E + 15) / 16;
    long nBlocks = (nTiles + 7) / 8;
    attn_logits_kernel<<<dim3((unsigned)nBlocks), dim3(256), 0, stream>>>(
        progF, voxF, src, dst, (const uint4*)w1bf, b1, W2, b2, logits, (int)nTiles, E);

    // 3. global softmax statistics
    reduce_max_partial<<<dim3(256), dim3(256), 0, stream>>>(logits, E, partials);
    reduce_max_final<<<dim3(1), dim3(256), 0, stream>>>(partials, &gstats[0]);
    reduce_sum_partial<<<dim3(256), dim3(256), 0, stream>>>(logits, E, &gstats[0], partials);
    reduce_sum_final<<<dim3(1), dim3(256), 0, stream>>>(partials, &gstats[1]);

    // 4. attention weights -> tail of d_out
    weights_kernel<<<dim3((unsigned)((E + 255) / 256)), dim3(256), 0, stream>>>(
        logits, E, gstats, outw);

    // 5. zero output_features, then scatter-add attended features
    long n4 = NVOX * (long)HID / 4;
    zero_kernel<<<dim3(2048), dim3(256), 0, stream>>>((float4*)out, n4);

    long nThreads = E * 32;
    scatter_kernel<<<dim3((unsigned)((nThreads + 255) / 256)), dim3(256), 0, stream>>>(
        progF, src, dst, outw, out, E);
}
